// Model_70068096467647
// MI455X (gfx1250) — compile-verified, matrix operand order checked by reading, not run on hardware
//
#include <hip/hip_runtime.h>
#include <hip/hip_bf16.h>

typedef _Float16 half_t;
typedef __attribute__((ext_vector_type(16))) _Float16 v16h;
typedef __attribute__((ext_vector_type(8)))  float    v8f;

#define LRELU_ALPHA 0.2f

__device__ __forceinline__ float lrelu(float x) { return x >= 0.f ? x : LRELU_ALPHA * x; }
__device__ __forceinline__ float sigm(float x)  { return 1.f / (1.f + __expf(-x)); }

__device__ __forceinline__ v8f wmma_f16(v16h a, v16h b, v8f c) {
    // D = A(16x32 f16) * B(32x16 f16) + C(16x16 f32)
    return __builtin_amdgcn_wmma_f32_16x16x32_f16(false, a, false, b, (short)0, c,
                                                  false, false);
}

// CDNA5 wave32 16-bit A/B fragment K index: VGPR j<4 -> K=2j+8*hi ; j>=4 -> K=16+2(j-4)+8*hi
__device__ __forceinline__ int frag_k(int j, int hi) {
    return (j < 4) ? (2 * j + 8 * hi) : (16 + 2 * (j - 4) + 8 * hi);
}

// A fragment (16M x 32K) from LDS f16, row-major with stride lda
__device__ __forceinline__ v16h load_a_lds(const half_t* A, int lda, int lane) {
    v16h a;
    int m = lane & 15, hi = (lane >> 4) & 1;
#pragma unroll
    for (int j = 0; j < 8; ++j) {
        int k = frag_k(j, hi);
        a[2 * j]     = A[m * lda + k];
        a[2 * j + 1] = A[m * lda + k + 1];
    }
    return a;
}

// A fragment from global f32 (convert to f16), row-major with stride lda
__device__ __forceinline__ v16h load_a_global_f32(const float* A, int lda, int lane) {
    v16h a;
    int m = lane & 15, hi = (lane >> 4) & 1;
#pragma unroll
    for (int j = 0; j < 8; ++j) {
        int k = frag_k(j, hi);
        a[2 * j]     = (half_t)A[m * lda + k];
        a[2 * j + 1] = (half_t)A[m * lda + k + 1];
    }
    return a;
}

// B fragment (32K x 16N) from global f32 row-major [K][N] with stride ldb (convert to f16)
__device__ __forceinline__ v16h load_b_global_f32(const float* B, int ldb, int lane) {
    v16h b;
    int n = lane & 15, hi = (lane >> 4) & 1;
#pragma unroll
    for (int j = 0; j < 8; ++j) {
        int k = frag_k(j, hi);
        b[2 * j]     = (half_t)B[k * ldb + n];
        b[2 * j + 1] = (half_t)B[(k + 1) * ldb + n];
    }
    return b;
}

// -------------------------------------------------------------------------
// Kernel 1: embedding + conv1d(width3, E->1) + maxpool(3) collapsed into
// three 128-entry vocab dot tables + LDS lookups.  p[b][t] = pooled conv out.
// -------------------------------------------------------------------------
__global__ __launch_bounds__(256) void conv_pool_kernel(
    const int*   __restrict__ inputs,  // [128][2048]
    const float* __restrict__ emb,     // [128][64]
    const float* __restrict__ conv_w,  // [3][64][1]
    const float* __restrict__ conv_b,  // [1]
    float*       __restrict__ p)       // [128][2048]
{
    const int T = 2048;
    __shared__ float dv[3][128];
    __shared__ int   idx[2048];

    int b = blockIdx.x, tid = threadIdx.x;

    for (int e = tid; e < 3 * 128; e += 256) {
        int w = e >> 7, v = e & 127;
        float s = 0.f;
#pragma unroll 8
        for (int k = 0; k < 64; ++k) s += emb[v * 64 + k] * conv_w[w * 64 + k];
        dv[w][v] = s;
    }
    for (int t = tid; t < T; t += 256) idx[t] = inputs[b * T + t];
    __syncthreads();

    float cb = conv_b[0];
    for (int t = tid; t < T; t += 256) {
        float y[3];
#pragma unroll
        for (int d = -1; d <= 1; ++d) {
            int tt = t + d;
            if (tt < 0 || tt >= T) { y[d + 1] = -__builtin_inff(); continue; }
            float s = cb;                       // zero-padded SAME conv
            if (tt - 1 >= 0) s += dv[0][idx[tt - 1]];
            s += dv[1][idx[tt]];
            if (tt + 1 < T)  s += dv[2][idx[tt + 1]];
            y[d + 1] = s;
        }
        p[b * T + t] = fmaxf(fmaxf(y[0], y[1]), y[2]);
    }
}

// -------------------------------------------------------------------------
// Kernel 2: LSTM scan.  8 blocks x 16 batch rows (one WMMA M-tile each).
// Recurrent kernel rk[64][256] lives as resident f16 B-fragments in VGPRs;
// h (f16) / c (f32) / z (f32) live in LDS.  Per step: 4 WMMA per wave.
// -------------------------------------------------------------------------
__global__ __launch_bounds__(256) void lstm_scan_kernel(
    const float* __restrict__ p,       // [128][2048] pooled conv output
    const float* __restrict__ lstm_k,  // [1][256]
    const float* __restrict__ lstm_rk, // [64][256]
    const float* __restrict__ lstm_b,  // [256]
    float*       __restrict__ hf)      // [128][64] final hidden state
{
    const int T = 2048;
    __shared__ half_t h_lds[16][64];
    __shared__ float  c_lds[16][64];
    __shared__ float  z_lds[16][256];
    __shared__ float  xk[256], xb[256];
    __shared__ float  pt[16];

    int tid = threadIdx.x, wv = tid >> 5, lane = tid & 31;
    int b0 = blockIdx.x * 16;
    int nl = lane & 15, hi = (lane >> 4) & 1;

    for (int e = tid; e < 16 * 64; e += 256) {
        h_lds[e >> 6][e & 63] = (half_t)0.f;
        c_lds[e >> 6][e & 63] = 0.f;
    }
    xk[tid] = lstm_k[tid];
    xb[tid] = lstm_b[tid];

    // Resident recurrent-kernel fragments: wave wv owns gate columns [32*wv, 32*wv+32)
    int col0 = wv * 32;
    v16h B00 = load_b_global_f32(lstm_rk + 0 * 256  + col0,      256, lane);
    v16h B01 = load_b_global_f32(lstm_rk + 32 * 256 + col0,      256, lane);
    v16h B10 = load_b_global_f32(lstm_rk + 0 * 256  + col0 + 16, 256, lane);
    v16h B11 = load_b_global_f32(lstm_rk + 32 * 256 + col0 + 16, 256, lane);
    __syncthreads();

    for (int t = 0; t < T; ++t) {
        if (tid < 16) {
            pt[tid] = p[(b0 + tid) * T + t];
            if ((t & 63) == 0 && t + 64 < T)
                __builtin_prefetch(&p[(b0 + tid) * T + t + 64], 0, 0);
        }
        __syncthreads();

        // z = h @ rk  (WMMA, f32 accumulate)  + p*k + b
        v16h a0 = load_a_lds(&h_lds[0][0],      64, lane);  // K 0..31
        v16h a1 = load_a_lds(&h_lds[0][0] + 32, 64, lane);  // K 32..63
        {
            v8f acc = {};
            acc = wmma_f16(a0, B00, acc);
            acc = wmma_f16(a1, B01, acc);
#pragma unroll
            for (int r = 0; r < 8; ++r) {
                int m = r + hi * 8, col = col0 + nl;
                z_lds[m][col] = acc[r] + pt[m] * xk[col] + xb[col];
            }
        }
        {
            v8f acc = {};
            acc = wmma_f16(a0, B10, acc);
            acc = wmma_f16(a1, B11, acc);
#pragma unroll
            for (int r = 0; r < 8; ++r) {
                int m = r + hi * 8, col = col0 + 16 + nl;
                z_lds[m][col] = acc[r] + pt[m] * xk[col] + xb[col];
            }
        }
        __syncthreads();

        // gate nonlinearity + state update (keras order i,f,c,o)
        for (int e = tid; e < 16 * 64; e += 256) {
            int m = e >> 6, u = e & 63;
            float ig = sigm(z_lds[m][u]);
            float fg = sigm(z_lds[m][64 + u]);
            float g  = lrelu(z_lds[m][128 + u]);
            float og = sigm(z_lds[m][192 + u]);
            float c  = fg * c_lds[m][u] + ig * g;
            c_lds[m][u] = c;
            h_lds[m][u] = (half_t)(og * lrelu(c));
        }
        __syncthreads();
    }

    for (int e = tid; e < 16 * 64; e += 256)
        hf[(b0 + (e >> 6)) * 64 + (e & 63)] = (float)h_lds[e >> 6][e & 63];
}

// -------------------------------------------------------------------------
// Kernel 3: head.  a1 = lrelu(hf @ w1 + b1) ; logits = a1 @ w2 + b2 ;
// row softmax.  Single block, 8 waves, all matmuls via WMMA.
// -------------------------------------------------------------------------
__global__ __launch_bounds__(256) void head_kernel(
    const float* __restrict__ hf,     // [128][64]
    const float* __restrict__ w1,     // [64][256]
    const float* __restrict__ b1,     // [256]
    const float* __restrict__ w2,     // [256][128]
    const float* __restrict__ b2,     // [128]
    float*       __restrict__ a1_ws,  // [128][256] scratch
    float*       __restrict__ lg_ws,  // [128][128] scratch
    float*       __restrict__ out)    // [128][128]
{
    int tid = threadIdx.x, wv = tid >> 5, lane = tid & 31;
    int nl = lane & 15, hi = (lane >> 4) & 1;

    // ---- GEMM1: M=128, N=256, K=64 ; wave wv owns columns [32*wv, 32*wv+32) ----
    int c0 = wv * 32;
    v16h B0a = load_b_global_f32(w1 + 0 * 256  + c0,      256, lane);
    v16h B0b = load_b_global_f32(w1 + 32 * 256 + c0,      256, lane);
    v16h B1a = load_b_global_f32(w1 + 0 * 256  + c0 + 16, 256, lane);
    v16h B1b = load_b_global_f32(w1 + 32 * 256 + c0 + 16, 256, lane);

    for (int mt = 0; mt < 8; ++mt) {
        const float* Ab = hf + mt * 16 * 64;
        v16h a0 = load_a_global_f32(Ab,      64, lane);
        v16h a1 = load_a_global_f32(Ab + 32, 64, lane);
        v8f acc0 = {}, acc1 = {};
        acc0 = wmma_f16(a0, B0a, acc0);
        acc0 = wmma_f16(a1, B0b, acc0);
        acc1 = wmma_f16(a0, B1a, acc1);
        acc1 = wmma_f16(a1, B1b, acc1);
#pragma unroll
        for (int r = 0; r < 8; ++r) {
            int m = mt * 16 + r + hi * 8;
            a1_ws[m * 256 + c0 + nl]      = lrelu(acc0[r] + b1[c0 + nl]);
            a1_ws[m * 256 + c0 + 16 + nl] = lrelu(acc1[r] + b1[c0 + 16 + nl]);
        }
    }
    __syncthreads();

    // ---- GEMM2: M=128, N=128, K=256 ; wave wv owns N-tile wv (16 cols) ----
    v16h Bw[8];
#pragma unroll
    for (int kt = 0; kt < 8; ++kt)
        Bw[kt] = load_b_global_f32(w2 + kt * 32 * 128 + wv * 16, 128, lane);

    for (int mt = 0; mt < 8; ++mt) {
        v8f acc = {};
#pragma unroll
        for (int kt = 0; kt < 8; ++kt) {
            v16h a = load_a_global_f32(a1_ws + mt * 16 * 256 + kt * 32, 256, lane);
            acc = wmma_f16(a, Bw[kt], acc);
        }
#pragma unroll
        for (int r = 0; r < 8; ++r) {
            int m = mt * 16 + r + hi * 8;
            lg_ws[m * 128 + wv * 16 + nl] = acc[r] + b2[wv * 16 + nl];
        }
    }
    __syncthreads();

    // ---- row softmax ----
    if (tid < 128) {
        const float* row = lg_ws + tid * 128;
        float mx = -__builtin_inff();
        for (int c = 0; c < 128; ++c) mx = fmaxf(mx, row[c]);
        float s = 0.f;
        for (int c = 0; c < 128; ++c) s += __expf(row[c] - mx);
        float inv = 1.f / s;
        for (int c = 0; c < 128; ++c) out[tid * 128 + c] = __expf(row[c] - mx) * inv;
    }
}

extern "C" void kernel_launch(void* const* d_in, const int* in_sizes, int n_in,
                              void* d_out, int out_size, void* d_ws, size_t ws_size,
                              hipStream_t stream) {
    (void)in_sizes; (void)n_in; (void)out_size; (void)ws_size;
    const int*   inputs  = (const int*)d_in[0];
    const float* emb     = (const float*)d_in[1];
    const float* conv_w  = (const float*)d_in[2];
    const float* conv_b  = (const float*)d_in[3];
    const float* lstm_k  = (const float*)d_in[4];
    const float* lstm_rk = (const float*)d_in[5];
    const float* lstm_b  = (const float*)d_in[6];
    const float* w1      = (const float*)d_in[7];
    const float* b1      = (const float*)d_in[8];
    const float* w2      = (const float*)d_in[9];
    const float* b2      = (const float*)d_in[10];
    float* out = (float*)d_out;

    float* ws  = (float*)d_ws;
    float* p   = ws;                  // 128*2048            = 262144 f32 (1 MB)
    float* hf  = ws + 262144;         // 128*64              = 8192   f32
    float* a1w = ws + 270336;         // 128*256             = 32768  f32
    float* lgw = ws + 303104;         // 128*128             = 16384  f32

    conv_pool_kernel<<<128, 256, 0, stream>>>(inputs, emb, conv_w, conv_b, p);
    lstm_scan_kernel<<<8, 256, 0, stream>>>(p, lstm_k, lstm_rk, lstm_b, hf);
    head_kernel<<<1, 256, 0, stream>>>(hf, w1, b1, w2, b2, a1w, lgw, out);
}
